// NerfRenderer_50843822850757
// MI455X (gfx1250) — matrix operand order repaired and checked by reading, not compiled, not run on hardware
//
#include <hip/hip_runtime.h>
#include <hip/hip_bf16.h>
#include <stdint.h>

typedef __attribute__((ext_vector_type(16))) _Float16 v16h;
typedef __attribute__((ext_vector_type(8)))  float    v8f;

#define N_RAYS   16384
#define HIDDEN   32
#define NC       192       // coarse samples (128 + 64)
#define NF       128       // fine (importance) samples
#define ALPHA_IMP 0.02f
#define INNER 1.0f
#define OUTER 2.0f

// flat output layout: image | weights | z_log_s | invdepth
#define OUT_IMG   0
#define OUT_W     (N_RAYS*3)
#define OUT_ZS    (OUT_W + N_RAYS*NF)
#define OUT_INVD  (OUT_ZS + N_RAYS*NF)

__device__ __forceinline__ float zlogc(int i){
  // segment 0: linspace(-1, 0 - 1/128, 128); segment 1: linspace(0, 1 - 1/64, 64)
  return (i < 128) ? (-1.0f + (float)i * (1.0f/128.0f))
                   : ((float)(i - 128) * (1.0f/64.0f));
}
__device__ __forceinline__ float dzlogc(int i){
  return (i < NC-1) ? (zlogc(i+1) - zlogc(i)) : 0.0f;
}
__device__ __forceinline__ float exp10f_(float x){ return __expf(x * 2.3025850929940457f); }
__device__ __forceinline__ float softplusf_(float x){
  return fmaxf(x, 0.0f) + log1pf(__expf(-fabsf(x)));
}
__device__ __forceinline__ float sigmoidf_(float x){ return 1.0f / (1.0f + __expf(-x)); }
__device__ __forceinline__ float rnd01(uint32_t a, uint32_t b){
  uint32_t x = a * 0x9E3779B9u ^ (b + 0x7F4A7C15u) * 0x85EBCA6Bu;
  x ^= x >> 16; x *= 0x7FEB352Du; x ^= x >> 15; x *= 0x846CA68Bu; x ^= x >> 16;
  return (float)(x >> 8) * (1.0f/16777216.0f);
}
__device__ __forceinline__ void mipscale(float& x, float& y, float& z){
  float dist = sqrtf(x*x + y*y + z*z);
  float s = (dist > INNER) ? ((OUTER - (OUTER - INNER)/dist)/dist) : 1.0f;
  x *= s; y *= s; z *= s;
}

// ISA §11 ds_swizzle group-of-32: offset = {xor_mask[14:10], or_mask[9:5], and_mask[4:0]}
// SWAPX16: xor=0x10, or=0, and=0x1f -> 0x401F  (swap lanes 0-15 <-> 16-31)
__device__ __forceinline__ float xor16f(float x){
  return __int_as_float(__builtin_amdgcn_ds_swizzle(__float_as_int(x), 0x401f));
}
__device__ __forceinline__ uint32_t xor16u(uint32_t x){
  return (uint32_t)__builtin_amdgcn_ds_swizzle((int)x, 0x401f);
}

union PackH { uint32_t u; _Float16 h[2]; };

__global__ __launch_bounds__(32)
void nerf_ray_kernel(const float* __restrict__ H, const float* __restrict__ W,
                     const float* __restrict__ Kmat, const float* __restrict__ E,
                     const float* __restrict__ BG,
                     const float* __restrict__ Wd1, const float* __restrict__ bd1,
                     const float* __restrict__ Wd2, const float* __restrict__ bd2,
                     const float* __restrict__ Wc1, const float* __restrict__ bc1,
                     const float* __restrict__ Wc2, const float* __restrict__ bc2,
                     float* __restrict__ out)
{
  __shared__ float s_sigma[NC];          // coarse sigmas (fine reuses first 128)
  __shared__ float s_buf[NC];            // coarse weights scratch
  __shared__ float s_cdf[NC];            // importance CDF
  __shared__ float s_zlogf[NF];          // fine z (log10)
  __shared__ float s_rgb[NF*3];          // fine rgb

  const int ray  = blockIdx.x;
  const int lane = threadIdx.x;          // 0..31 (wave32)
  const int g    = lane >> 4;            // half-wave id (K-half for A/B layouts)
  const int r    = lane & 15;            // row (A) / column = point (B,D)

  // ---- resident A fragments: weights transposed, neurons on the M side ----
  // A layout: lane holds row M = r; element e -> K = e + 16*g
  v16h Ad1f0, Ad1f1, Ac1f0, Ac1f1;
  #pragma unroll
  for (int e = 0; e < 16; e++){
    int K = e + 16*g;
    float w0 = (K < 3) ? Wd1[K*HIDDEN + r]        : 0.0f;  // neurons 0..15
    float w1 = (K < 3) ? Wd1[K*HIDDEN + 16 + r]   : 0.0f;  // neurons 16..31
    Ad1f0[e] = (_Float16)w0;
    Ad1f1[e] = (_Float16)w1;
    Ac1f0[e] = (_Float16)Wc1[K*HIDDEN + r];                // hfeat rows of Wc1
    Ac1f1[e] = (_Float16)Wc1[K*HIDDEN + 16 + r];
  }

  // ---- per-lane per-neuron scalars (D layout: VGPR v -> neuron v+8g / 16+v+8g) --
  float bd1A[8], bd1B[8], wd2A[8], wd2B[8];
  float wc2A[8][3], wc2B[8][3];
  #pragma unroll
  for (int v = 0; v < 8; v++){
    int mA = v + 8*g, mB = 16 + v + 8*g;
    bd1A[v] = bd1[mA]; bd1B[v] = bd1[mB];
    wd2A[v] = Wd2[mA]; wd2B[v] = Wd2[mB];
    #pragma unroll
    for (int c = 0; c < 3; c++){
      wc2A[v][c] = Wc2[mA*3 + c];
      wc2B[v][c] = Wc2[mB*3 + c];
    }
  }
  const float bd2v = bd2[0];
  float bc2v[3], bgv[3];
  #pragma unroll
  for (int c = 0; c < 3; c++){ bc2v[c] = bc2[c]; bgv[c] = BG[c]; }

  // ---- ray geometry: cam_dirs = dirs @ inv(K)^T ; rays_d = E[:3,:3] @ cam ----
  float k00=Kmat[0],k01=Kmat[1],k02=Kmat[2],k10=Kmat[3],k11=Kmat[4],k12=Kmat[5],
        k20=Kmat[6],k21=Kmat[7],k22=Kmat[8];
  float det = k00*(k11*k22-k12*k21) - k01*(k10*k22-k12*k20) + k02*(k10*k21-k11*k20);
  float id  = 1.0f/det;
  float i00=(k11*k22-k12*k21)*id, i01=(k02*k21-k01*k22)*id, i02=(k01*k12-k02*k11)*id;
  float i10=(k12*k20-k10*k22)*id, i11=(k00*k22-k02*k20)*id, i12=(k02*k10-k00*k12)*id;
  float i20=(k10*k21-k11*k20)*id, i21=(k01*k20-k00*k21)*id, i22=(k00*k11-k01*k10)*id;
  float dx = W[ray] + 0.5f, dy = H[ray] + 0.5f;
  float cxd = i00*dx + i01*dy + i02;
  float cyd = i10*dx + i11*dy + i12;
  float czd = i20*dx + i21*dy + i22;
  const float* Er = E + (size_t)ray*16;
  float rdx = Er[0]*cxd + Er[1]*cyd + Er[2]*czd;
  float rdy = Er[4]*cxd + Er[5]*cyd + Er[6]*czd;
  float rdz = Er[8]*cxd + Er[9]*cyd + Er[10]*czd;
  float rox = Er[3], roy = Er[7], roz = Er[11];
  float rlen = sqrtf(rdx*rdx + rdy*rdy + rdz*rdz);
  float ndx = rdx/rlen, ndy = rdy/rlen, ndz = rdz/rlen;
  // fold direction rows of Wc1 into per-neuron color bias (dirs constant per ray)
  float bcolA[8], bcolB[8];
  #pragma unroll
  for (int v = 0; v < 8; v++){
    int mA = v + 8*g, mB = 16 + v + 8*g;
    bcolA[v] = bc1[mA] + ndx*Wc1[32*HIDDEN+mA] + ndy*Wc1[33*HIDDEN+mA] + ndz*Wc1[34*HIDDEN+mA];
    bcolB[v] = bc1[mB] + ndx*Wc1[32*HIDDEN+mB] + ndy*Wc1[33*HIDDEN+mB] + ndz*Wc1[34*HIDDEN+mB];
  }

  // ================= coarse pass: 12 tiles of 16 points ======================
  for (int t = 0; t < 12; t++){
    int p = t*16 + r;
    float z  = exp10f_(zlogc(p));
    float px = rox + rdx*z, py = roy + rdy*z, pz = roz + rdz*z;
    mipscale(px, py, pz);

    // B fragment: points^T (K = coords padded to 32, N = point). K = e + 16*g.
    v16h b;
    #pragma unroll
    for (int e = 0; e < 16; e++) b[e] = (_Float16)0.0f;
    b[0] = (_Float16)((g == 0) ? px : 0.0f);
    b[1] = (_Float16)((g == 0) ? py : 0.0f);
    b[2] = (_Float16)((g == 0) ? pz : 0.0f);

    v8f c0, c1;
    #pragma unroll
    for (int e = 0; e < 8; e++){ c0[e] = 0.0f; c1[e] = 0.0f; }
    c0 = __builtin_amdgcn_wmma_f32_16x16x32_f16(false, Ad1f0, false, b, (short)0, c0, false, false);
    c1 = __builtin_amdgcn_wmma_f32_16x16x32_f16(false, Ad1f1, false, b, (short)0, c1, false, false);

    // sigma = softplus( hfeat . Wd2 + bd2 ), lane-local over 16 neurons + 1 swap
    float part = 0.0f;
    #pragma unroll
    for (int v = 0; v < 8; v++){
      part += fmaxf(c0[v] + bd1A[v], 0.0f) * wd2A[v];
      part += fmaxf(c1[v] + bd1B[v], 0.0f) * wd2B[v];
    }
    part += xor16f(part);
    // both half-waves hold the same value: unconditional duplicate store is benign
    s_sigma[t*16 + r] = softplusf_(part + bd2v);
  }
  __syncthreads();

  // ---- transmittance weights: w_i = (1-e^{-s_i d_i}) e^{-prefix} (6/lane) ----
  {
    float lexc[6]; float csum = 0.0f;
    #pragma unroll
    for (int k = 0; k < 6; k++){
      int i = lane*6 + k;
      float s = s_sigma[i]*dzlogc(i);
      lexc[k] = csum; csum += s;
    }
    float inc = csum;
    #pragma unroll
    for (int off = 1; off < 32; off <<= 1){
      float o = __shfl_up(inc, off);
      if (lane >= off) inc += o;
    }
    float base = inc - csum;
    #pragma unroll
    for (int k = 0; k < 6; k++){
      int i = lane*6 + k;
      float s = s_sigma[i]*dzlogc(i);
      s_buf[i] = (1.0f - __expf(-s)) * __expf(-(base + lexc[k]));
    }
  }
  __syncthreads();

  // ---- reweight (max filter + floor + segment scale), normalized CDF ----
  {
    float wr[6]; float lsum = 0.0f;
    #pragma unroll
    for (int k = 0; k < 6; k++){
      int i = lane*6 + k;
      float wm = s_buf[i];
      float wl = (i > 0)     ? s_buf[i-1] : 0.0f;
      float wn = (i < NC-1)  ? s_buf[i+1] : 0.0f;
      float v  = 0.5f*(fmaxf(wl, wm) + fmaxf(wm, wn)) + ALPHA_IMP/(float)NC;
      v *= (i < 128) ? (128.0f/192.0f) : (64.0f/192.0f);
      wr[k] = v; lsum += v;
    }
    float tot = lsum;
    tot += xor16f(tot);
    #pragma unroll
    for (int m = 8; m >= 1; m >>= 1) tot += __shfl_xor(tot, m);
    float lex[6]; float cc = 0.0f;
    #pragma unroll
    for (int k = 0; k < 6; k++){ lex[k] = cc; cc += wr[k]; }
    float inc = cc;
    #pragma unroll
    for (int off = 1; off < 32; off <<= 1){
      float o = __shfl_up(inc, off);
      if (lane >= off) inc += o;
    }
    float base = inc - cc;
    float invtot = 1.0f/tot;
    #pragma unroll
    for (int k = 0; k < 6; k++){
      int i = lane*6 + k;
      s_cdf[i] = (base + lex[k] + wr[k]) * invtot;   // inclusive cumsum
    }
  }
  __syncthreads();

  // ---- stratified inverse-CDF sampling (4 samples/lane) ----
  {
    float cdf1 = s_cdf[1], cdfm = s_cdf[190];
    #pragma unroll
    for (int k = 0; k < 4; k++){
      int js = lane*4 + k;
      float u = ((float)js + rnd01((uint32_t)ray, (uint32_t)js)) * (1.0f/(float)NF);
      u = u*(cdfm - cdf1) + cdf1;
      int lo = 0, hi = 189;                          // searchsorted(cdf[1:190], u, 'right')
      while (lo < hi){
        int mid = (lo + hi) >> 1;
        if (s_cdf[1 + mid] <= u) lo = mid + 1; else hi = mid;
      }
      int ind = lo + 1;                              // 1..190
      float cb = s_cdf[ind-1], ca = s_cdf[ind];
      float zb = 0.5f*(zlogc(ind-1) + zlogc(ind));
      float za = 0.5f*(zlogc(ind)   + zlogc(ind+1));
      float tt = (u - cb) / (ca - cb);
      float zf = zb + (za - zb)*tt;
      s_zlogf[js] = zf;
      out[OUT_ZS + (size_t)ray*NF + js] = (zf + 1.0f)*0.5f;   // (z - lo)/(hi - lo)
    }
  }
  __syncthreads();

  // ================= fine pass: 8 tiles of 16 points =========================
  for (int t = 0; t < 8; t++){
    int p = t*16 + r;
    float zl = s_zlogf[p];
    float z  = exp10f_(zl);
    float px = rox + rdx*z, py = roy + rdy*z, pz = roz + rdz*z;
    mipscale(px, py, pz);

    v16h b;
    #pragma unroll
    for (int e = 0; e < 16; e++) b[e] = (_Float16)0.0f;
    b[0] = (_Float16)((g == 0) ? px : 0.0f);
    b[1] = (_Float16)((g == 0) ? py : 0.0f);
    b[2] = (_Float16)((g == 0) ? pz : 0.0f);

    v8f c0, c1;
    #pragma unroll
    for (int e = 0; e < 8; e++){ c0[e] = 0.0f; c1[e] = 0.0f; }
    c0 = __builtin_amdgcn_wmma_f32_16x16x32_f16(false, Ad1f0, false, b, (short)0, c0, false, false);
    c1 = __builtin_amdgcn_wmma_f32_16x16x32_f16(false, Ad1f1, false, b, (short)0, c1, false, false);

    // hfeat for this lane's point: neurons v+8g (h0) and 16+v+8g (h1)
    float h0[8], h1[8];
    #pragma unroll
    for (int v = 0; v < 8; v++){
      h0[v] = fmaxf(c0[v] + bd1A[v], 0.0f);
      h1[v] = fmaxf(c1[v] + bd1B[v], 0.0f);
    }
    // sigma (32->1): lane-local dot + 1 swap
    float part = 0.0f;
    #pragma unroll
    for (int v = 0; v < 8; v++) part += h0[v]*wd2A[v] + h1[v]*wd2B[v];
    part += xor16f(part);
    s_sigma[t*16 + r] = softplusf_(part + bd2v);

    // build B fragment of hfeat (K = neuron = e + 16*g, N = point).
    // Each half only needs the other half's values: pre-select what to send
    // (g0 sends h1, g1 sends h0), convert to f16 and pack pairs -> 4 DS swaps.
    uint32_t rp[4];
    #pragma unroll
    for (int v = 0; v < 4; v++){
      PackH pk;
      pk.h[0] = (_Float16)((g == 0) ? h1[2*v]   : h0[2*v]);
      pk.h[1] = (_Float16)((g == 0) ? h1[2*v+1] : h0[2*v+1]);
      rp[v] = xor16u(pk.u);
    }
    v16h b2;
    #pragma unroll
    for (int e = 0; e < 8; e++){
      PackH pr; pr.u = rp[e >> 1];
      _Float16 rv = pr.h[e & 1];
      // g0: K=e     -> own h0[e];   K=8+e  -> h0[e] of lane^16 (= rv)
      // g1: K=16+e  -> h1[e] of lane^16 (= rv);  K=24+e -> own h1[e]
      b2[e]     = (g == 0) ? (_Float16)h0[e] : rv;
      b2[8 + e] = (g == 0) ? rv : (_Float16)h1[e];
    }

    v8f d0, d1;
    #pragma unroll
    for (int e = 0; e < 8; e++){ d0[e] = 0.0f; d1[e] = 0.0f; }
    d0 = __builtin_amdgcn_wmma_f32_16x16x32_f16(false, Ac1f0, false, b2, (short)0, d0, false, false);
    d1 = __builtin_amdgcn_wmma_f32_16x16x32_f16(false, Ac1f1, false, b2, (short)0, d1, false, false);

    float g0[8], g1[8];
    #pragma unroll
    for (int v = 0; v < 8; v++){
      g0[v] = fmaxf(d0[v] + bcolA[v], 0.0f);
      g1[v] = fmaxf(d1[v] + bcolB[v], 0.0f);
    }
    // rgb (32->3): lane-local dots + 1 swap per channel
    #pragma unroll
    for (int c = 0; c < 3; c++){
      float pc = 0.0f;
      #pragma unroll
      for (int v = 0; v < 8; v++) pc += g0[v]*wc2A[v][c] + g1[v]*wc2B[v][c];
      pc += xor16f(pc);
      s_rgb[(t*16 + r)*3 + c] = sigmoidf_(pc + bc2v[c]);
    }
  }
  __syncthreads();

  // ---- fine weights + composite (4 samples/lane) ----
  {
    float lexc[4]; float csum = 0.0f;
    #pragma unroll
    for (int k = 0; k < 4; k++){
      int i = lane*4 + k;
      float dzl = (i < NF-1) ? (s_zlogf[i+1] - s_zlogf[i]) : 0.0f;
      float s = s_sigma[i]*dzl;
      lexc[k] = csum; csum += s;
    }
    float inc = csum;
    #pragma unroll
    for (int off = 1; off < 32; off <<= 1){
      float o = __shfl_up(inc, off);
      if (lane >= off) inc += o;
    }
    float base = inc - csum;
    float img0 = 0.f, img1 = 0.f, img2 = 0.f, invd = 0.f, wsum = 0.f;
    #pragma unroll
    for (int k = 0; k < 4; k++){
      int i = lane*4 + k;
      float dzl = (i < NF-1) ? (s_zlogf[i+1] - s_zlogf[i]) : 0.0f;
      float s = s_sigma[i]*dzl;
      float wgt = (1.0f - __expf(-s)) * __expf(-(base + lexc[k]));
      out[OUT_W + (size_t)ray*NF + i] = wgt;
      img0 += wgt * s_rgb[i*3 + 0];
      img1 += wgt * s_rgb[i*3 + 1];
      img2 += wgt * s_rgb[i*3 + 2];
      invd += wgt / exp10f_(s_zlogf[i]);
      wsum += wgt;
    }
    img0 += xor16f(img0); img1 += xor16f(img1); img2 += xor16f(img2);
    invd += xor16f(invd); wsum += xor16f(wsum);
    #pragma unroll
    for (int m = 8; m >= 1; m >>= 1){
      img0 += __shfl_xor(img0, m); img1 += __shfl_xor(img1, m); img2 += __shfl_xor(img2, m);
      invd += __shfl_xor(invd, m); wsum += __shfl_xor(wsum, m);
    }
    if (lane == 0){
      float rem = 1.0f - wsum;
      out[OUT_IMG + (size_t)ray*3 + 0] = img0 + rem*bgv[0];
      out[OUT_IMG + (size_t)ray*3 + 1] = img1 + rem*bgv[1];
      out[OUT_IMG + (size_t)ray*3 + 2] = img2 + rem*bgv[2];
      out[OUT_INVD + ray] = invd;
    }
  }
}

extern "C" void kernel_launch(void* const* d_in, const int* in_sizes, int n_in,
                              void* d_out, int out_size, void* d_ws, size_t ws_size,
                              hipStream_t stream)
{
  (void)in_sizes; (void)n_in; (void)out_size; (void)d_ws; (void)ws_size;
  const float* H   = (const float*)d_in[1];
  const float* W   = (const float*)d_in[2];
  const float* Km  = (const float*)d_in[3];
  const float* E   = (const float*)d_in[4];
  const float* BG  = (const float*)d_in[5];
  const float* Wd1 = (const float*)d_in[6];
  const float* bd1 = (const float*)d_in[7];
  const float* Wd2 = (const float*)d_in[8];
  const float* bd2 = (const float*)d_in[9];
  const float* Wc1 = (const float*)d_in[10];
  const float* bc1 = (const float*)d_in[11];
  const float* Wc2 = (const float*)d_in[12];
  const float* bc2 = (const float*)d_in[13];

  nerf_ray_kernel<<<dim3(N_RAYS), dim3(32), 0, stream>>>(
      H, W, Km, E, BG, Wd1, bd1, Wd2, bd2, Wc1, bc1, Wc2, bc2, (float*)d_out);
}
